// InvariantPointAttention_19413252178133
// MI455X (gfx1250) — matrix operand order repaired and checked by reading, not compiled, not run on hardware
//
#include <hip/hip_runtime.h>
#include <hip/hip_bf16.h>

typedef __attribute__((ext_vector_type(16))) _Float16 v16h;
typedef __attribute__((ext_vector_type(8)))  _Float16 v8h;
typedef __attribute__((ext_vector_type(8)))  float    v8f;

#define NTOK 1024
#define NHEAD 12
#define CDIM 384
#define YW 1008          // 192+192+192+144+144+144
#define FEAT 32          // 16 scalar + 12 point + 1 ones + 3 pad
#define FW 256           // 240 output features padded to 256
#define INV_SQRT_DK 0.25f
#define CPT 0.058925565f // gamma * inv_sqrt_dk, gamma = sqrt(2/36)

// ---------- WMMA fragment loaders (16x16x32 f16) ----------
// A-matrix 16x32: lane L -> row (L&15); K = k0 + 16h .. k0 + 16h + 15, h = L>>4
__device__ __forceinline__ v16h load_afrag(const _Float16* base, int row0, int ld,
                                           int k0, int lane) {
  int half = lane >> 4, m = lane & 15;
  const _Float16* p = base + (size_t)(row0 + m) * ld + k0;
  union { v16h v; v8h h[2]; } u;
  u.h[0] = *(const v8h*)(p + 16 * half);
  u.h[1] = *(const v8h*)(p + 16 * half + 8);
  return u.v;
}
// B-matrix 32x16 from an N-major (transposed) array: lane L -> col (L&15);
// K = k0 + 16h .. k0 + 16h + 15 (contiguous 32B)
__device__ __forceinline__ v16h load_bfrag(const _Float16* baseT, int col0, int ld,
                                           int k0, int lane) {
  int half = lane >> 4, n = lane & 15;
  return *(const v16h*)(baseT + (size_t)(col0 + n) * ld + k0 + 16 * half);
}

// LDS byte offset of a generic pointer to __shared__ (aperture maps addr[31:0])
__device__ __forceinline__ unsigned ldsoff(const void* p) {
  return (unsigned)(unsigned long long)p;
}

// ---------- kernel 0: pack f32 -> f16, transpose weights ----------
__global__ void pack_kernel(const float* __restrict__ x,
                            const float* __restrict__ Wq, const float* __restrict__ Wk,
                            const float* __restrict__ Wv, const float* __restrict__ Wqp,
                            const float* __restrict__ Wkp, const float* __restrict__ Wvp,
                            const float* __restrict__ Wout,
                            _Float16* __restrict__ X16, _Float16* __restrict__ WtT,
                            _Float16* __restrict__ WoutT) {
  const int T1 = NTOK * CDIM;          // X16
  const int T2 = YW * CDIM;            // WtT (1008 x 384)
  const int T3 = CDIM * FW;            // WoutT (384 x 256)
  int idx = blockIdx.x * blockDim.x + threadIdx.x;
  if (idx < T1) {
    X16[idx] = (_Float16)x[idx];
  } else if (idx < T1 + T2) {
    int j = idx - T1, n = j / CDIM, k = j % CDIM;
    float v;
    if      (n < 192)  v = Wq [k * 192 + n];
    else if (n < 384)  v = Wk [k * 192 + (n - 192)];
    else if (n < 576)  v = Wv [k * 192 + (n - 384)];
    else if (n < 720)  v = Wqp[k * 144 + (n - 576)];
    else if (n < 864)  v = Wkp[k * 144 + (n - 720)];
    else               v = Wvp[k * 144 + (n - 864)];
    WtT[(size_t)n * CDIM + k] = (_Float16)v;
  } else if (idx < T1 + T2 + T3) {
    int j = idx - T1 - T2, n = j / FW, k = j % FW;
    WoutT[(size_t)n * FW + k] = (_Float16)(k < 240 ? Wout[k * CDIM + n] : 0.0f);
  }
}

// ---------- kernel 1: projection GEMM 1024x1008x384 ----------
__global__ void __launch_bounds__(256) gemm_proj(const _Float16* __restrict__ A,
                                                 const _Float16* __restrict__ BT,
                                                 float* __restrict__ C) {
  int lane = threadIdx.x & 31, wid = threadIdx.x >> 5;
  int task = blockIdx.x * 8 + wid;            // 4032 tasks = 64 x 63 tiles
  int m0 = (task / 63) * 16, n0 = (task % 63) * 16;
  int half = lane >> 4, ln = lane & 15;
  v8f acc = {};
  for (int kk = 0; kk < CDIM; kk += 32) {
    v16h a = load_afrag(A, m0, CDIM, kk, lane);
    v16h b = load_bfrag(BT, n0, CDIM, kk, lane);
    acc = __builtin_amdgcn_wmma_f32_16x16x32_f16(false, a, false, b, (short)0, acc,
                                                 false, false);
  }
#pragma unroll
  for (int r = 0; r < 8; ++r)
    C[(size_t)(m0 + r + 8 * half) * YW + n0 + ln] = acc[r];
}

// ---------- kernel 2: build fused attention features ----------
__global__ void feat_kernel(const float* __restrict__ Y, const float* __restrict__ rot,
                            const float* __restrict__ trans,
                            _Float16* __restrict__ Qf, _Float16* __restrict__ Kf,
                            _Float16* __restrict__ VfT, float* __restrict__ bias) {
  int tid = blockIdx.x * blockDim.x + threadIdx.x;
  if (tid >= NTOK * NHEAD) return;
  int h = tid % NHEAD, i = tid / NHEAD;
  const float* y = Y + (size_t)i * YW;
  const float* R = rot + i * 9;
  const float* t = trans + i * 3;
  _Float16* q = Qf + ((size_t)h * NTOK + i) * FEAT;
  _Float16* k = Kf + ((size_t)h * NTOK + i) * FEAT;
#pragma unroll
  for (int d = 0; d < 16; ++d) {
    q[d] = (_Float16)(y[h * 16 + d] * INV_SQRT_DK);
    k[d] = (_Float16)(y[192 + h * 16 + d]);
  }
  float k2 = 0.f;
#pragma unroll
  for (int p = 0; p < 4; ++p) {
    float q0 = y[576 + h * 12 + p * 3 + 0], q1 = y[576 + h * 12 + p * 3 + 1],
          q2 = y[576 + h * 12 + p * 3 + 2];
    float p0 = y[720 + h * 12 + p * 3 + 0], p1 = y[720 + h * 12 + p * 3 + 1],
          p2 = y[720 + h * 12 + p * 3 + 2];
#pragma unroll
    for (int c = 0; c < 3; ++c) {
      float qg = R[c * 3] * q0 + R[c * 3 + 1] * q1 + R[c * 3 + 2] * q2 + t[c];
      float kg = R[c * 3] * p0 + R[c * 3 + 1] * p1 + R[c * 3 + 2] * p2 + t[c];
      q[16 + p * 3 + c] = (_Float16)(qg * CPT);
      k[16 + p * 3 + c] = (_Float16)kg;
      k2 += kg * kg;
    }
  }
#pragma unroll
  for (int d = 28; d < 32; ++d) { q[d] = (_Float16)0.f; k[d] = (_Float16)0.f; }
  bias[h * NTOK + i] = -0.5f * CPT * k2;

  _Float16* vT = VfT + (size_t)h * FEAT * NTOK;
#pragma unroll
  for (int d = 0; d < 16; ++d) vT[(size_t)d * NTOK + i] = (_Float16)(y[384 + h * 16 + d]);
#pragma unroll
  for (int p = 0; p < 4; ++p) {
    float v0 = y[864 + h * 12 + p * 3 + 0], v1 = y[864 + h * 12 + p * 3 + 1],
          v2 = y[864 + h * 12 + p * 3 + 2];
#pragma unroll
    for (int c = 0; c < 3; ++c) {
      float vg = R[c * 3] * v0 + R[c * 3 + 1] * v1 + R[c * 3 + 2] * v2 + t[c];
      vT[(size_t)(16 + p * 3 + c) * NTOK + i] = (_Float16)vg;
    }
  }
  vT[(size_t)28 * NTOK + i] = (_Float16)1.0f;   // ones column -> softmax denominator
#pragma unroll
  for (int d = 29; d < 32; ++d) vT[(size_t)d * NTOK + i] = (_Float16)0.f;
}

// ---------- kernel 3: flash attention ----------
// block = 8 waves = 128 rows of ONE head; K/V tiles double-buffered in LDS via
// global_load_async_to_lds_b128 (ASYNCcnt), shared by all 8 waves.
__global__ void __launch_bounds__(256) attn_kernel(const _Float16* __restrict__ Qf,
                                                   const _Float16* __restrict__ Kf,
                                                   const _Float16* __restrict__ VfT,
                                                   const float* __restrict__ bias,
                                                   float* __restrict__ Of) {
  __shared__ __align__(32) _Float16 kbuf[2][64 * 32];   // [j_local][feat]
  __shared__ __align__(32) _Float16 vbuf[2][32 * 64];   // [feat][j_local]
  __shared__ __align__(32) _Float16 plds[8][16 * 64];   // per-wave P transpose

  int tid = threadIdx.x;
  int lane = tid & 31, wid = tid >> 5;
  int head = blockIdx.x >> 3;                    // 12 heads x 8 blocks
  int i0 = (blockIdx.x & 7) * 128 + wid * 16;
  int half = lane >> 4, ln = lane & 15;

  const _Float16* Qh = Qf + (size_t)head * NTOK * FEAT;
  const _Float16* Kh = Kf + (size_t)head * NTOK * FEAT;
  const _Float16* Vh = VfT + (size_t)head * FEAT * NTOK;
  const float*    bh = bias + head * NTOK;

  unsigned long long kbase = (unsigned long long)Kh;   // uniform per block
  unsigned long long vbase = (unsigned long long)Vh;
  int vf = tid >> 3, vj = tid & 7;                     // V-tile copy mapping

  v16h qfrag = load_afrag(Qh, i0, FEAT, 0, lane);
  v8f oA = {}, oB = {};
  float mrow[8];
#pragma unroll
  for (int r = 0; r < 8; ++r) mrow[r] = -3.0e38f;

  // stage tile 0 (K: 64 rows x 64B contiguous; V: 32 feature rows x 128B)
  {
    unsigned kdst = ldsoff(&kbuf[0][0]) + tid * 16;
    unsigned koff = tid * 16;
    asm volatile("global_load_async_to_lds_b128 %0, %1, %2"
                 :: "v"(kdst), "v"(koff), "s"(kbase) : "memory");
    unsigned vdst = ldsoff(&vbuf[0][0]) + (vf * 64 + vj * 8) * 2;
    unsigned voff = (vf * NTOK + vj * 8) * 2;
    asm volatile("global_load_async_to_lds_b128 %0, %1, %2"
                 :: "v"(vdst), "v"(voff), "s"(vbase) : "memory");
  }

  for (int it = 0; it < 16; ++it) {
    int j0 = it * 64;
    int st = it & 1;
    asm volatile("s_wait_asynccnt 0x0" ::: "memory");
    __syncthreads();                       // tile `it` resident in all lanes' view
    if (it + 1 < 16) {                     // prefetch next tile into other buffer
      int jn = j0 + 64, sn = st ^ 1;
      unsigned kdst = ldsoff(&kbuf[sn][0]) + tid * 16;
      unsigned koff = jn * 64 + tid * 16;
      asm volatile("global_load_async_to_lds_b128 %0, %1, %2"
                   :: "v"(kdst), "v"(koff), "s"(kbase) : "memory");
      unsigned vdst = ldsoff(&vbuf[sn][0]) + (vf * 64 + vj * 8) * 2;
      unsigned voff = (vf * NTOK + jn + vj * 8) * 2;
      asm volatile("global_load_async_to_lds_b128 %0, %1, %2"
                   :: "v"(vdst), "v"(voff), "s"(vbase) : "memory");
    }

    // ---- S = Qf x Kf^T for 16 rows x 64 cols (4 WMMAs), K-frags from LDS ----
    v8f s[4];
#pragma unroll
    for (int t = 0; t < 4; ++t) {
      v16h bk = *(const v16h*)&kbuf[st][(t * 16 + ln) * 32 + 16 * half];
      v8f z = {};
      s[t] = __builtin_amdgcn_wmma_f32_16x16x32_f16(false, qfrag, false, bk, (short)0,
                                                    z, false, false);
    }
    // bias + tile-wide max (online softmax is exact for any row-uniform shift)
    float vmax = -3.0e38f;
#pragma unroll
    for (int t = 0; t < 4; ++t) {
      float bt = bh[j0 + t * 16 + ln];
#pragma unroll
      for (int r = 0; r < 8; ++r) {
        s[t][r] += bt;
        vmax = fmaxf(vmax, s[t][r]);
      }
    }
    vmax = fmaxf(vmax, __shfl_xor(vmax, 1, 32));
    vmax = fmaxf(vmax, __shfl_xor(vmax, 2, 32));
    vmax = fmaxf(vmax, __shfl_xor(vmax, 4, 32));
    vmax = fmaxf(vmax, __shfl_xor(vmax, 8, 32));
    vmax = fmaxf(vmax, __shfl_xor(vmax, 16, 32));
#pragma unroll
    for (int r = 0; r < 8; ++r) {
      float newm = fmaxf(mrow[r], vmax);
      float sc = __expf(mrow[r] - newm);
      mrow[r] = newm;
      oA[r] *= sc;
      oB[r] *= sc;                          // ones-column rescales with everything
      int row = r + 8 * half;
#pragma unroll
      for (int t = 0; t < 4; ++t) {
        float p = __expf(s[t][r] - newm);
        plds[wid][row * 64 + t * 16 + ln] = (_Float16)p;
      }
    }
    // ---- O += P x V (P transposed to A-frag layout via per-wave LDS) ----
    union { v16h v; v8h h[2]; } u0, u1;
    const _Float16* pb = &plds[wid][ln * 64];
    u0.h[0] = *(const v8h*)(pb + 16 * half);
    u0.h[1] = *(const v8h*)(pb + 8 + 16 * half);
    u1.h[0] = *(const v8h*)(pb + 32 + 16 * half);
    u1.h[1] = *(const v8h*)(pb + 40 + 16 * half);
    v16h bv00 = *(const v16h*)&vbuf[st][ln * 64 + 16 * half];          // f0..15,j0..31
    v16h bv01 = *(const v16h*)&vbuf[st][ln * 64 + 32 + 16 * half];     // f0..15,j32..63
    v16h bv10 = *(const v16h*)&vbuf[st][(16 + ln) * 64 + 16 * half];   // f16..31
    v16h bv11 = *(const v16h*)&vbuf[st][(16 + ln) * 64 + 32 + 16 * half];
    oA = __builtin_amdgcn_wmma_f32_16x16x32_f16(false, u0.v, false, bv00, (short)0, oA,
                                                false, false);
    oA = __builtin_amdgcn_wmma_f32_16x16x32_f16(false, u1.v, false, bv01, (short)0, oA,
                                                false, false);
    oB = __builtin_amdgcn_wmma_f32_16x16x32_f16(false, u0.v, false, bv10, (short)0, oB,
                                                false, false);
    oB = __builtin_amdgcn_wmma_f32_16x16x32_f16(false, u1.v, false, bv11, (short)0, oB,
                                                false, false);
  }

  // denominator = ones-column (feature 28 -> oB lane 12 of own half)
#pragma unroll
  for (int r = 0; r < 8; ++r) {
    float denom = __shfl(oB[r], (lane & 16) + 12, 32);
    float inv = 1.0f / denom;
    int tok = i0 + r + 8 * half;
    size_t base = ((size_t)tok * NHEAD + head) * FEAT;
    Of[base + ln]      = oA[r] * inv;
    Of[base + 16 + ln] = oB[r] * inv;
  }
}

// ---------- kernel 4: norms + concat (||R^T c|| == ||c||, R orthonormal) ----------
__global__ void post_kernel(const float* __restrict__ Of, const float* __restrict__ trans,
                            _Float16* __restrict__ F16) {
  int tid = blockIdx.x * blockDim.x + threadIdx.x;
  if (tid >= NTOK * NHEAD) return;
  int h = tid % NHEAD, i = tid / NHEAD;
  const float* o = Of + ((size_t)i * NHEAD + h) * FEAT;
  const float* t = trans + i * 3;
  _Float16* f = F16 + (size_t)i * FW;
#pragma unroll
  for (int d = 0; d < 16; ++d) f[h * 16 + d] = (_Float16)o[d];
#pragma unroll
  for (int p = 0; p < 4; ++p) {
    float g0 = o[16 + p * 3 + 0] - t[0];
    float g1 = o[16 + p * 3 + 1] - t[1];
    float g2 = o[16 + p * 3 + 2] - t[2];
    f[192 + h * 4 + p] = (_Float16)sqrtf(g0 * g0 + g1 * g1 + g2 * g2);
  }
  if (h == 0) {
#pragma unroll
    for (int d = 240; d < 256; ++d) f[d] = (_Float16)0.f;
  }
}

// ---------- kernel 5: output GEMM 1024x384x256 + bias ----------
__global__ void __launch_bounds__(256) gemm_out(const _Float16* __restrict__ A,
                                                const _Float16* __restrict__ BT,
                                                const float* __restrict__ bout,
                                                float* __restrict__ C) {
  int lane = threadIdx.x & 31, wid = threadIdx.x >> 5;
  int task = blockIdx.x * 8 + wid;            // 1536 tasks = 64 x 24 tiles
  int m0 = (task / 24) * 16, n0 = (task % 24) * 16;
  int half = lane >> 4, ln = lane & 15;
  v8f acc = {};
  for (int kk = 0; kk < FW; kk += 32) {
    v16h a = load_afrag(A, m0, FW, kk, lane);
    v16h b = load_bfrag(BT, n0, FW, kk, lane);
    acc = __builtin_amdgcn_wmma_f32_16x16x32_f16(false, a, false, b, (short)0, acc,
                                                 false, false);
  }
  float bb = bout[n0 + ln];
#pragma unroll
  for (int r = 0; r < 8; ++r)
    C[(size_t)(m0 + r + 8 * half) * CDIM + n0 + ln] = acc[r] + bb;
}

extern "C" void kernel_launch(void* const* d_in, const int* in_sizes, int n_in,
                              void* d_out, int out_size, void* d_ws, size_t ws_size,
                              hipStream_t stream) {
  const float* x     = (const float*)d_in[0];
  const float* rot   = (const float*)d_in[1];
  const float* trans = (const float*)d_in[2];
  const float* Wq    = (const float*)d_in[3];
  const float* Wk    = (const float*)d_in[4];
  const float* Wv    = (const float*)d_in[5];
  const float* Wqp   = (const float*)d_in[6];
  const float* Wkp   = (const float*)d_in[7];
  const float* Wvp   = (const float*)d_in[8];
  const float* Wout  = (const float*)d_in[9];
  const float* bout  = (const float*)d_in[10];
  float* out = (float*)d_out;

  uint8_t* ws = (uint8_t*)d_ws;
  size_t off = 0;
  auto alloc = [&](size_t bytes) {
    void* p = ws + off;
    off = (off + bytes + 255) & ~(size_t)255;
    return p;
  };
  _Float16* X16   = (_Float16*)alloc((size_t)NTOK * CDIM * 2);
  _Float16* WtT   = (_Float16*)alloc((size_t)YW * CDIM * 2);
  _Float16* WoutT = (_Float16*)alloc((size_t)CDIM * FW * 2);
  float*    Y     = (float*)   alloc((size_t)NTOK * YW * 4);
  _Float16* Qf    = (_Float16*)alloc((size_t)NHEAD * NTOK * FEAT * 2);
  _Float16* Kf    = (_Float16*)alloc((size_t)NHEAD * NTOK * FEAT * 2);
  _Float16* VfT   = (_Float16*)alloc((size_t)NHEAD * FEAT * NTOK * 2);
  float*    bias  = (float*)   alloc((size_t)NHEAD * NTOK * 4);
  float*    Of    = (float*)   alloc((size_t)NTOK * NHEAD * FEAT * 4);
  _Float16* F16   = (_Float16*)alloc((size_t)NTOK * FW * 2);

  const int packN = NTOK * CDIM + YW * CDIM + CDIM * FW;
  pack_kernel<<<(packN + 255) / 256, 256, 0, stream>>>(x, Wq, Wk, Wv, Wqp, Wkp, Wvp,
                                                       Wout, X16, WtT, WoutT);
  gemm_proj<<<(64 * 63) / 8, 256, 0, stream>>>(X16, WtT, Y);
  feat_kernel<<<(NTOK * NHEAD) / 256, 256, 0, stream>>>(Y, rot, trans, Qf, Kf, VfT, bias);
  attn_kernel<<<NHEAD * 8, 256, 0, stream>>>(Qf, Kf, VfT, bias, Of);
  post_kernel<<<(NTOK * NHEAD) / 256, 256, 0, stream>>>(Of, trans, F16);
  gemm_out<<<(64 * 24) / 8, 256, 0, stream>>>(F16, WoutT, bout, out);
}
